// model_feature1_39719857553500
// MI455X (gfx1250) — matrix operand order repaired and checked by reading, not compiled, not run on hardware
//
#include <hip/hip_runtime.h>

typedef __attribute__((ext_vector_type(16))) _Float16 v16h;
typedef __attribute__((ext_vector_type(8)))  float    v8f;

#define EMDIM 128
#define MI_NODES 16384
#define DIS_NODES 8192
#define CDIV(a,b) (((a)+(b)-1)/(b))

// ---------------- small helpers ----------------
__device__ __forceinline__ float eluf(float x)  { return x > 0.f ? x : (expf(x) - 1.f); }
__device__ __forceinline__ float lrelu(float x) { return x >= 0.f ? x : 0.2f * x; }

// monotonic float <-> u32 mapping (order-preserving for all finite floats)
__device__ __forceinline__ unsigned fenc(float f) {
  unsigned u = __float_as_uint(f);
  return (u & 0x80000000u) ? ~u : (u | 0x80000000u);
}
__device__ __forceinline__ float fdec(unsigned u) {
  return __uint_as_float((u & 0x80000000u) ? (u ^ 0x80000000u) : ~u);
}

// ---------------- WMMA 16x16 tile (K=128), A: f32 row-major, Bt: f16 row-major [N x 128] ----------------
// A fragment (16x32, contiguous k order matching B): lane m=lane&15, hi=lane>>4:
//   a[0..15] = A[m][k0 + hi*16 .. +15]
// B fragment (32x16): lane n=lane&15, hi: b[i] = B[k0 + hi*16 + i][n] = Bt[n][k0 + hi*16 + i]
__device__ __forceinline__ v8f tile_accum(const float* __restrict__ A,
                                          const _Float16* __restrict__ Bt,
                                          int rb, int cb, int lane) {
  const int m  = lane & 15;
  const int hi = lane >> 4;
  v8f c = {};
#pragma unroll
  for (int k0 = 0; k0 < EMDIM; k0 += 32) {
    const float* pa = A + (size_t)(rb + m) * EMDIM + k0 + hi * 16;
    v16h a;
#pragma unroll
    for (int i = 0; i < 8; ++i) a[i]     = (_Float16)pa[i];
#pragma unroll
    for (int i = 0; i < 8; ++i) a[8 + i] = (_Float16)pa[8 + i];

    const _Float16* pb = Bt + (size_t)(cb + m) * EMDIM + k0 + hi * 16;
    v16h b;
#pragma unroll
    for (int i = 0; i < 16; ++i) b[i] = pb[i];

    c = __builtin_amdgcn_wmma_f32_16x16x32_f16(false, a, false, b, (short)0, c,
                                               false, false);
  }
  return c;
}

// ---------------- elementwise / init kernels ----------------
__global__ void k_fill_f32(float* p, float v, int n) {
  int i = blockIdx.x * blockDim.x + threadIdx.x;
  if (i < n) p[i] = v;
}
__global__ void k_init_minmax(unsigned* mm) {
  if (threadIdx.x == 0) { mm[0] = 0xFFFFFFFFu; mm[1] = 0u; }
}
__global__ void k_deg(float* deg, const int* __restrict__ col,
                      const float* __restrict__ ew, int E) {
  int e = blockIdx.x * blockDim.x + threadIdx.x;
  if (e < E) atomicAdd(&deg[col[e]], ew ? ew[e] : 1.0f);
}
__global__ void k_rsqrt(float* p, int n) {
  int i = blockIdx.x * blockDim.x + threadIdx.x;
  if (i < n) p[i] = rsqrtf(p[i]);
}
__global__ void k_w_t_half(const float* __restrict__ W, _Float16* __restrict__ Wt) {
  int i = blockIdx.x * blockDim.x + threadIdx.x;  // 128*128 threads
  if (i < EMDIM * EMDIM) {
    int k = i >> 7, n = i & 127;
    Wt[n * EMDIM + k] = (_Float16)W[k * EMDIM + n];
  }
}

// ---------------- feature GEMM: C[M x 128] = A[M x 128] @ Wt^T ----------------
__global__ void k_gemm_feat(const float* __restrict__ A, const _Float16* __restrict__ Bt,
                            float* __restrict__ C) {
  int lane  = threadIdx.x;
  int tileN = blockIdx.x * blockDim.y + threadIdx.y;  // 0..7
  int tileM = blockIdx.y;
  v8f c = tile_accum(A, Bt, tileM * 16, tileN * 16, lane);
  int hi = lane >> 4, n = lane & 15;
  int row0 = tileM * 16 + hi * 8;
  int cn   = tileN * 16 + n;
#pragma unroll
  for (int r = 0; r < 8; ++r) C[(size_t)(row0 + r) * EMDIM + cn] = c[r];
}

// ---------------- GCN kernels ----------------
__global__ void k_gcn_self(const float* __restrict__ h, const float* __restrict__ dinv,
                           const float* __restrict__ b, float* __restrict__ out, int n) {
  int i = blockIdx.x * blockDim.x + threadIdx.x;
  if (i >= n * EMDIM) return;
  int node = i >> 7, d = i & 127;
  float di = dinv[node];
  out[i] = di * di * h[i] + b[d];
}
__global__ void k_gcn_edge(const float* __restrict__ h, float* __restrict__ out,
                           const int* __restrict__ row, const int* __restrict__ col,
                           const float* __restrict__ ew, const float* __restrict__ dinv,
                           int E) {
  int gid = blockIdx.x * blockDim.x + threadIdx.x;
  int e = gid >> 5, lane = gid & 31;
  if (e >= E) return;
  int r = row[e], c = col[e];
  float w = ew ? ew[e] : 1.0f;
  float norm = dinv[r] * w * dinv[c];
  float4 v = ((const float4*)(h + (size_t)r * EMDIM))[lane];
  float* op = out + (size_t)c * EMDIM + lane * 4;
  atomicAdd(op + 0, norm * v.x);
  atomicAdd(op + 1, norm * v.y);
  atomicAdd(op + 2, norm * v.z);
  atomicAdd(op + 3, norm * v.w);
}
__global__ void k_elu_acc(const float* __restrict__ in, float* __restrict__ cur,
                          float* __restrict__ fsum, int count) {
  int i = blockIdx.x * blockDim.x + threadIdx.x;
  if (i < count) {
    float v = eluf(in[i]);
    cur[i] = v;
    fsum[i] += v;
  }
}

// ---------------- GAT kernels ----------------
__global__ void k_attn(const float* __restrict__ h, const float* __restrict__ asrc,
                       const float* __restrict__ adst, float* __restrict__ es,
                       float* __restrict__ ed, int n) {
  int gid = blockIdx.x * blockDim.x + threadIdx.x;
  int node = gid >> 5, lane = gid & 31;
  if (node >= n) return;
  const float* hp = h + (size_t)node * EMDIM;
  float s = 0.f, d = 0.f;
#pragma unroll
  for (int i = 0; i < 4; ++i) {
    float hv = hp[lane + 32 * i];
    s += hv * asrc[lane + 32 * i];
    d += hv * adst[lane + 32 * i];
  }
#pragma unroll
  for (int off = 16; off; off >>= 1) {
    s += __shfl_down(s, off);
    d += __shfl_down(d, off);
  }
  if (lane == 0) { es[node] = s; ed[node] = d; }
}
__global__ void k_gat_selfmax(const float* es, const float* ed, unsigned* menc, int n) {
  int i = blockIdx.x * blockDim.x + threadIdx.x;
  if (i < n) menc[i] = fenc(lrelu(es[i] + ed[i]));
}
__global__ void k_gat_edgemax(const float* __restrict__ es, const float* __restrict__ ed,
                              const int* __restrict__ row, const int* __restrict__ col,
                              unsigned* menc, int E) {
  int e = blockIdx.x * blockDim.x + threadIdx.x;
  if (e < E) atomicMax(&menc[col[e]], fenc(lrelu(es[row[e]] + ed[col[e]])));
}
__global__ void k_gat_selfsum(const float* es, const float* ed, const unsigned* menc,
                              float* ssum, int n) {
  int i = blockIdx.x * blockDim.x + threadIdx.x;
  if (i < n) ssum[i] = expf(lrelu(es[i] + ed[i]) - fdec(menc[i]));
}
__global__ void k_gat_edgesum(const float* __restrict__ es, const float* __restrict__ ed,
                              const int* __restrict__ row, const int* __restrict__ col,
                              const unsigned* __restrict__ menc, float* ssum, int E) {
  int e = blockIdx.x * blockDim.x + threadIdx.x;
  if (e < E) {
    int c = col[e];
    atomicAdd(&ssum[c], expf(lrelu(es[row[e]] + ed[c]) - fdec(menc[c])));
  }
}
__global__ void k_gat_self(const float* __restrict__ h, const float* __restrict__ es,
                           const float* __restrict__ ed, const unsigned* __restrict__ menc,
                           const float* __restrict__ ssum, const float* __restrict__ b,
                           float* __restrict__ out, int n) {
  int i = blockIdx.x * blockDim.x + threadIdx.x;
  if (i >= n * EMDIM) return;
  int node = i >> 7, d = i & 127;
  float a = expf(lrelu(es[node] + ed[node]) - fdec(menc[node])) / ssum[node];
  out[i] = a * h[i] + b[d];
}
__global__ void k_gat_edgeagg(const float* __restrict__ h, float* __restrict__ out,
                              const int* __restrict__ row, const int* __restrict__ col,
                              const float* __restrict__ es, const float* __restrict__ ed,
                              const unsigned* __restrict__ menc,
                              const float* __restrict__ ssum, int E) {
  int gid = blockIdx.x * blockDim.x + threadIdx.x;
  int e = gid >> 5, lane = gid & 31;
  if (e >= E) return;
  int r = row[e], c = col[e];
  float alpha = expf(lrelu(es[r] + ed[c]) - fdec(menc[c])) / ssum[c];
  float4 v = ((const float4*)(h + (size_t)r * EMDIM))[lane];
  float* op = out + (size_t)c * EMDIM + lane * 4;
  atomicAdd(op + 0, alpha * v.x);
  atomicAdd(op + 1, alpha * v.y);
  atomicAdd(op + 2, alpha * v.z);
  atomicAdd(op + 3, alpha * v.w);
}
__global__ void k_finalize(const float* __restrict__ fsum, float* __restrict__ fout,
                           _Float16* __restrict__ hout, int count) {
  int i = blockIdx.x * blockDim.x + threadIdx.x;
  if (i < count) {
    float v = fsum[i] * (1.0f / 3.0f);
    fout[i] = v;
    if (hout) hout[i] = (_Float16)v;
  }
}

// ---------------- score GEMM: pass 1 (min/max only), pass 2 (normalize+store) ----------------
__global__ void k_score_minmax(const float* __restrict__ mi, const _Float16* __restrict__ dis16,
                               unsigned* __restrict__ mm) {
  int lane  = threadIdx.x;
  int tileN = blockIdx.x * blockDim.y + threadIdx.y;
  int tileM = blockIdx.y;
  v8f c = tile_accum(mi, dis16, tileM * 16, tileN * 16, lane);
  float mn = c[0], mx = c[0];
#pragma unroll
  for (int r = 1; r < 8; ++r) { mn = fminf(mn, c[r]); mx = fmaxf(mx, c[r]); }
#pragma unroll
  for (int off = 16; off; off >>= 1) {
    mn = fminf(mn, __shfl_xor(mn, off));
    mx = fmaxf(mx, __shfl_xor(mx, off));
  }
  if (lane == 0) {
    atomicMin(&mm[0], fenc(mn));
    atomicMax(&mm[1], fenc(mx));
  }
}
__global__ void k_score_norm(const float* __restrict__ mi, const _Float16* __restrict__ dis16,
                             const unsigned* __restrict__ mm, float* __restrict__ score) {
  int lane  = threadIdx.x;
  int tileN = blockIdx.x * blockDim.y + threadIdx.y;
  int tileM = blockIdx.y;
  v8f c = tile_accum(mi, dis16, tileM * 16, tileN * 16, lane);
  float mn = fdec(mm[0]), mx = fdec(mm[1]);
  bool ok = mx > mn;
  float inv = ok ? 1.0f / (mx - mn) : 0.0f;
  int hi = lane >> 4, n0 = tileN * 16 + (lane & 15);
  int row0 = tileM * 16 + hi * 8;
#pragma unroll
  for (int r = 0; r < 8; ++r) {
    float v = ok ? (c[r] - mn) * inv : 0.0f;
    v = fminf(fmaxf(v, 0.0f), 1.0f);
    // streaming 537MB output, never re-read: non-temporal store (TH=NT) so it
    // does not displace the L2-resident GEMM operands used by both passes.
    __builtin_nontemporal_store(v, &score[(size_t)(row0 + r) * DIS_NODES + n0]);
  }
}

// ---------------- host-side orchestration ----------------
// workspace layout (bytes)
static const size_t OFF_A     = 0;                        // cur features   16384*128*4
static const size_t OFF_B     = OFF_A + 8388608;          // agg/out
static const size_t OFF_H     = OFF_B + 8388608;          // h = x@w
static const size_t OFF_S     = OFF_H + 8388608;          // fea_sum
static const size_t OFF_DINV1 = OFF_S + 8388608;          // 16384*4
static const size_t OFF_DINV2 = OFF_DINV1 + 65536;
static const size_t OFF_ES    = OFF_DINV2 + 65536;
static const size_t OFF_ED    = OFF_ES + 65536;
static const size_t OFF_MENC  = OFF_ED + 65536;
static const size_t OFF_SSUM  = OFF_MENC + 65536;
static const size_t OFF_DIS16 = OFF_SSUM + 65536;         // 8192*128*2
static const size_t OFF_WT    = OFF_DIS16 + 2097152;      // 128*128*2
static const size_t OFF_MM    = OFF_WT + 32768;           // 2 u32

static void run_graph(const float* x0, const float* w1, const float* b1,
                      const float* w2, const float* b2, const float* w3,
                      const float* asrc, const float* adst, const float* b3,
                      const float* ew, const int* edges, int n, int E,
                      float* fea_out, _Float16* fea_h16, char* ws, hipStream_t s) {
  const int* row = edges;
  const int* col = edges + E;
  float*    bufA  = (float*)(ws + OFF_A);
  float*    bufB  = (float*)(ws + OFF_B);
  float*    h     = (float*)(ws + OFF_H);
  float*    fsum  = (float*)(ws + OFF_S);
  float*    dinv1 = (float*)(ws + OFF_DINV1);
  float*    dinv2 = (float*)(ws + OFF_DINV2);
  float*    es    = (float*)(ws + OFF_ES);
  float*    ed    = (float*)(ws + OFF_ED);
  unsigned* menc  = (unsigned*)(ws + OFF_MENC);
  float*    ssum  = (float*)(ws + OFF_SSUM);
  _Float16* wt    = (_Float16*)(ws + OFF_WT);

  const int nf = n * EMDIM;
  dim3 gemmGrid(2, n / 16), gemmBlk(32, 4);

  // degree / norm factors (self-loop weight 1 folded into init)
  k_fill_f32<<<CDIV(n, 256), 256, 0, s>>>(dinv1, 1.0f, n);
  k_deg<<<CDIV(E, 256), 256, 0, s>>>(dinv1, col, ew, E);
  k_rsqrt<<<CDIV(n, 256), 256, 0, s>>>(dinv1, n);
  k_fill_f32<<<CDIV(n, 256), 256, 0, s>>>(dinv2, 1.0f, n);
  k_deg<<<CDIV(E, 256), 256, 0, s>>>(dinv2, col, nullptr, E);
  k_rsqrt<<<CDIV(n, 256), 256, 0, s>>>(dinv2, n);
  k_fill_f32<<<CDIV(nf, 256), 256, 0, s>>>(fsum, 0.0f, nf);

  // layer 1: GCN with edge weights
  k_w_t_half<<<CDIV(EMDIM * EMDIM, 256), 256, 0, s>>>(w1, wt);
  k_gemm_feat<<<gemmGrid, gemmBlk, 0, s>>>(x0, wt, h);
  k_gcn_self<<<CDIV(nf, 256), 256, 0, s>>>(h, dinv1, b1, bufB, n);
  k_gcn_edge<<<CDIV(E * 32, 256), 256, 0, s>>>(h, bufB, row, col, ew, dinv1, E);
  k_elu_acc<<<CDIV(nf, 256), 256, 0, s>>>(bufB, bufA, fsum, nf);

  // layer 2: GCN, unit edge weights
  k_w_t_half<<<CDIV(EMDIM * EMDIM, 256), 256, 0, s>>>(w2, wt);
  k_gemm_feat<<<gemmGrid, gemmBlk, 0, s>>>(bufA, wt, h);
  k_gcn_self<<<CDIV(nf, 256), 256, 0, s>>>(h, dinv2, b2, bufB, n);
  k_gcn_edge<<<CDIV(E * 32, 256), 256, 0, s>>>(h, bufB, row, col, nullptr, dinv2, E);
  k_elu_acc<<<CDIV(nf, 256), 256, 0, s>>>(bufB, bufA, fsum, nf);

  // layer 3: GAT (heads=1)
  k_w_t_half<<<CDIV(EMDIM * EMDIM, 256), 256, 0, s>>>(w3, wt);
  k_gemm_feat<<<gemmGrid, gemmBlk, 0, s>>>(bufA, wt, h);
  k_attn<<<CDIV(n * 32, 256), 256, 0, s>>>(h, asrc, adst, es, ed, n);
  k_gat_selfmax<<<CDIV(n, 256), 256, 0, s>>>(es, ed, menc, n);
  k_gat_edgemax<<<CDIV(E, 256), 256, 0, s>>>(es, ed, row, col, menc, E);
  k_gat_selfsum<<<CDIV(n, 256), 256, 0, s>>>(es, ed, menc, ssum, n);
  k_gat_edgesum<<<CDIV(E, 256), 256, 0, s>>>(es, ed, row, col, menc, ssum, E);
  k_gat_self<<<CDIV(nf, 256), 256, 0, s>>>(h, es, ed, menc, ssum, b3, bufB, n);
  k_gat_edgeagg<<<CDIV(E * 32, 256), 256, 0, s>>>(h, bufB, row, col, es, ed, menc, ssum, E);
  k_elu_acc<<<CDIV(nf, 256), 256, 0, s>>>(bufB, bufA, fsum, nf);

  k_finalize<<<CDIV(nf, 256), 256, 0, s>>>(fsum, fea_out, fea_h16, nf);
}

extern "C" void kernel_launch(void* const* d_in, const int* in_sizes, int n_in,
                              void* d_out, int out_size, void* d_ws, size_t ws_size,
                              hipStream_t stream) {
  (void)in_sizes; (void)n_in; (void)out_size; (void)ws_size;
  const float* mm_x0   = (const float*)d_in[0];
  const float* dd_x0   = (const float*)d_in[1];
  const float* w1_mm   = (const float*)d_in[2];
  const float* b1_mm   = (const float*)d_in[3];
  const float* w2_mm   = (const float*)d_in[4];
  const float* b2_mm   = (const float*)d_in[5];
  const float* w3_mm   = (const float*)d_in[6];
  const float* asrc_mm = (const float*)d_in[7];
  const float* adst_mm = (const float*)d_in[8];
  const float* b3_mm   = (const float*)d_in[9];
  const float* w1_dd   = (const float*)d_in[10];
  const float* b1_dd   = (const float*)d_in[11];
  const float* w2_dd   = (const float*)d_in[12];
  const float* b2_dd   = (const float*)d_in[13];
  const float* w3_dd   = (const float*)d_in[14];
  const float* asrc_dd = (const float*)d_in[15];
  const float* adst_dd = (const float*)d_in[16];
  const float* b3_dd   = (const float*)d_in[17];
  const float* mm_ew   = (const float*)d_in[18];
  const float* dd_ew   = (const float*)d_in[19];
  const int*   mm_edges = (const int*)d_in[20];
  const int*   dd_edges = (const int*)d_in[21];

  const int E_MM = MI_NODES * 32;
  const int E_DD = DIS_NODES * 32;

  char* ws = (char*)d_ws;
  float* score   = (float*)d_out;                              // [16384 x 8192]
  float* mi_fea  = score + (size_t)MI_NODES * DIS_NODES;       // [16384 x 128]
  float* dis_fea = mi_fea + (size_t)MI_NODES * EMDIM;          // [8192 x 128]
  _Float16* dis16 = (_Float16*)(ws + OFF_DIS16);
  unsigned* mmx   = (unsigned*)(ws + OFF_MM);

  run_graph(mm_x0, w1_mm, b1_mm, w2_mm, b2_mm, w3_mm, asrc_mm, adst_mm, b3_mm,
            mm_ew, mm_edges, MI_NODES, E_MM, mi_fea, nullptr, ws, stream);
  run_graph(dd_x0, w1_dd, b1_dd, w2_dd, b2_dd, w3_dd, asrc_dd, adst_dd, b3_dd,
            dd_ew, dd_edges, DIS_NODES, E_DD, dis_fea, dis16, ws, stream);

  // score = mi_fea @ dis_fea^T, two WMMA passes (min/max, then normalize+store)
  k_init_minmax<<<1, 32, 0, stream>>>(mmx);
  dim3 sGrid(DIS_NODES / 64, MI_NODES / 16), sBlk(32, 4);
  k_score_minmax<<<sGrid, sBlk, 0, stream>>>(mi_fea, dis16, mmx);
  k_score_norm<<<sGrid, sBlk, 0, stream>>>(mi_fea, dis16, mmx, score);
}
